// OrthoLinear_53609781788904
// MI455X (gfx1250) — matrix operand order repaired and checked by reading, not compiled, not run on hardware
//
#include <hip/hip_runtime.h>

// ---- problem constants (match reference) ----
#define IN_F        8192
#define OUT_F       8192
#define NROWS       512          // 8 * 64 batch rows
#define NNZ_PER_ROW 409
#define KSTEP       32           // WMMA K per step (f16)
#define MTILE       64           // output features per block
#define NWAVE       32           // batch rows per wave (16 waves -> 512)
#define BLOCK       512          // 16 waves

typedef __attribute__((ext_vector_type(16))) _Float16 v16h;
typedef __attribute__((ext_vector_type(8)))  _Float16 v8h;
typedef __attribute__((ext_vector_type(4)))  _Float16 v4h;
typedef __attribute__((ext_vector_type(8)))  float    v8f;

union AFrag { v16h v; v8h h[2]; };

// ------------------------------------------------------------------
// Kernel 1: convert x (fp32) -> fp16 into workspace. 8 elems/thread.
// ------------------------------------------------------------------
__global__ __launch_bounds__(256) void x_to_f16_kernel(
    const float* __restrict__ x, _Float16* __restrict__ xh) {
  size_t i = ((size_t)blockIdx.x * blockDim.x + threadIdx.x) * 8;
  float4 a = *(const float4*)(x + i);
  float4 b = *(const float4*)(x + i + 4);
  v8h h;
  h[0] = (_Float16)a.x; h[1] = (_Float16)a.y;
  h[2] = (_Float16)a.z; h[3] = (_Float16)a.w;
  h[4] = (_Float16)b.x; h[5] = (_Float16)b.y;
  h[6] = (_Float16)b.z; h[7] = (_Float16)b.w;
  *(v8h*)(xh + i) = h;
}

// ------------------------------------------------------------------
// Kernel 2: fused int4-dequant + WMMA GEMM.
//   out[n, o] = scale[o] * sum_k xh[n,k] * (nib(base[o,k]) - 8)
// Block: 512 threads (16 waves), owns M=64 output features and all
// 512 batch rows -> the 128 MB packed-weight stream is read from HBM
// exactly once. Per k-step the block cooperatively dequants a 64x32
// weight tile into double-buffered LDS (each element converted
// exactly once, amortized over 16 waves); each wave then computes a
// 64x32 output tile with 4 A-frags x 2 B-frags = 8 v_wmma per step.
// x (8 MB fp16) stays L2-resident. C = 64 VGPRs/lane -> ~2 waves/SIMD.
// ------------------------------------------------------------------
__global__ __launch_bounds__(BLOCK) void gemm_dq_kernel(
    const int*      __restrict__ base,    // [OUT_F][IN_F/2] one packed byte per int32
    const float*    __restrict__ scales,  // [OUT_F]
    const _Float16* __restrict__ xh,      // [NROWS][IN_F]
    float*          __restrict__ out) {   // [NROWS][OUT_F]
  __shared__ _Float16 As[2][MTILE * KSTEP];  // 64 rows x 32 k fp16, double buffered (8 KB)

  const int tid  = threadIdx.x;
  const int lane = tid & 31;
  const int wave = tid >> 5;               // 0..15 -> n range [32w, 32w+32)
  const int o0   = blockIdx.x * MTILE;

  // dequant assignment: thread -> (weight row, 2 consecutive packed ints)
  const int drow = tid >> 3;               // 0..63
  const int dk2  = tid & 7;                // 2-int group -> k sub-range of 4
  const int* brow = base + (size_t)(o0 + drow) * (IN_F / 2) + dk2 * 2;

  // A-fragment addressing (contiguous K, matching B):
  //   lanes 0-15 : M=lane,    K = 0..15
  //   lanes 16-31: M=lane-16, K = 16..31
  const int am = lane & 15;
  const int ak = (lane >> 4) * 16;

  // B-fragment addressing: column = n, 16 contiguous K per lane.
  const _Float16* xb =
      xh + (size_t)(wave * NWAVE + (lane & 15)) * IN_F + (lane >> 4) * 16;

  v8f c[4][2] = {};

  for (int k0 = 0, it = 0; k0 < IN_F; k0 += KSTEP, ++it) {
    const int p = it & 1;

    // --- cooperative dequant of 64x32 weight tile into LDS ---
    int2 pk = *(const int2*)&brow[k0 >> 1];
    __builtin_prefetch(&brow[(k0 >> 1) + 128], 0, 3);  // ~8 steps ahead, near cache
    v4h h;
    h[0] = (_Float16)(( pk.x       & 0xF) - 8);
    h[1] = (_Float16)(((pk.x >> 4) & 0xF) - 8);
    h[2] = (_Float16)(( pk.y       & 0xF) - 8);
    h[3] = (_Float16)(((pk.y >> 4) & 0xF) - 8);
    *(v4h*)&As[p][drow * KSTEP + dk2 * 4] = h;
    __syncthreads();

    // --- 4 A fragments from LDS (shared by all 16 waves) ---
    AFrag a[4];
#pragma unroll
    for (int g = 0; g < 4; ++g) {
      const int arow = (16 * g + am) * KSTEP;
      a[g].h[0] = *(const v8h*)&As[p][arow + ak];        // K block 0
      a[g].h[1] = *(const v8h*)&As[p][arow + ak + 8];    // K block 1
    }

    // --- 2 B fragments from x (L2-resident) ---
    v16h b[2];
#pragma unroll
    for (int j = 0; j < 2; ++j)
      b[j] = *(const v16h*)(xb + k0 + (size_t)(16 * j) * IN_F);

    // --- 8 WMMAs: 64x32 wave tile ---
#pragma unroll
    for (int g = 0; g < 4; ++g)
#pragma unroll
      for (int j = 0; j < 2; ++j)
        c[g][j] = __builtin_amdgcn_wmma_f32_16x16x32_f16(
            false, a[g].v, false, b[j], (short)0, c[g][j], false, false);
  }

  // --- apply per-output-row scale and store ---
  // C layout: VGPR r holds (M = r + 8*(lane>=16), N = lane&15)
  const int nlo = lane & 15;
  const int mhi = (lane >> 4) * 8;
#pragma unroll
  for (int g = 0; g < 4; ++g) {
#pragma unroll
    for (int r = 0; r < 8; ++r) {
      const int m = 16 * g + r + mhi;
      const float s = scales[o0 + m];
      const size_t ocol = (size_t)(o0 + m);
#pragma unroll
      for (int j = 0; j < 2; ++j) {
        const int n = wave * NWAVE + 16 * j + nlo;
        out[(size_t)n * OUT_F + ocol] = c[g][j][r] * s;
      }
    }
  }
}

// ------------------------------------------------------------------
// Kernel 3: CSR residual  out[n,o] += ALPHA * sum_t val[t]*x[n,col[t]]
// One block per output row o; vals/indices staged in LDS; each thread
// owns distinct n -> plain read-modify-write, no atomics needed.
// ------------------------------------------------------------------
__global__ __launch_bounds__(256) void sparse_add_kernel(
    const float*    __restrict__ x,     // [NROWS][IN_F] fp32
    const _Float16* __restrict__ vals,  // [NNZ]
    const int*      __restrict__ cols,  // [NNZ]
    const int*      __restrict__ ptr,   // [OUT_F+1]
    float*          __restrict__ out) { // [NROWS][OUT_F]
  __shared__ float sv[NNZ_PER_ROW];
  __shared__ int   si[NNZ_PER_ROW];

  const int o     = blockIdx.x;
  const int start = ptr[o];

  for (int t = threadIdx.x; t < NNZ_PER_ROW; t += 256) {
    sv[t] = (float)vals[start + t];
    si[t] = cols[start + t];
  }
  __syncthreads();

  for (int n = threadIdx.x; n < NROWS; n += 256) {
    const float* xr = x + (size_t)n * IN_F;
    float acc = 0.0f;
#pragma unroll 4
    for (int t = 0; t < NNZ_PER_ROW; ++t)
      acc += sv[t] * xr[si[t]];
    out[(size_t)n * OUT_F + o] += acc;   // ALPHA == 1.0
  }
}

// ------------------------------------------------------------------
extern "C" void kernel_launch(void* const* d_in, const int* in_sizes, int n_in,
                              void* d_out, int out_size, void* d_ws, size_t ws_size,
                              hipStream_t stream) {
  const float*    x      = (const float*)   d_in[0];  // (8,64,8192) fp32
  const int*      base   = (const int*)     d_in[1];  // (8192,4096) int32 (packed byte each)
  const float*    scales = (const float*)   d_in[2];  // (8192,1)
  const _Float16* vals   = (const _Float16*)d_in[3];  // (NNZ,) fp16
  const int*      cols   = (const int*)     d_in[4];  // (NNZ,)
  const int*      ptr    = (const int*)     d_in[5];  // (8193,)
  float*          out    = (float*)d_out;             // (8,64,8192) fp32

  // workspace: x in fp16, 512*8192*2 = 8 MB
  _Float16* xh = (_Float16*)d_ws;

  // 1) fp32 -> fp16 conversion of x
  const int total = NROWS * IN_F;                 // 4,194,304
  x_to_f16_kernel<<<total / (256 * 8), 256, 0, stream>>>(x, xh);

  // 2) fused dequant + WMMA GEMM (128 blocks x 16 waves, M=64 each)
  gemm_dq_kernel<<<OUT_F / MTILE, BLOCK, 0, stream>>>(base, scales, xh, out);

  // 3) sparse residual accumulation
  sparse_add_kernel<<<OUT_F, 256, 0, stream>>>(x, vals, cols, ptr, out);
}
